// LabelGCNAttentionRNNv5_25744033972574
// MI455X (gfx1250) — matrix operand order repaired and checked by reading, not compiled, not run on hardware
//
#include <hip/hip_runtime.h>
#include <hip/hip_bf16.h>
#include <stdint.h>

#define NN_NODES  10000
#define NN_EDGES  160000
#define DHID      512
#define NN_LABELS 8192

typedef __attribute__((ext_vector_type(16))) __bf16        bf16x16;
typedef __attribute__((ext_vector_type(8)))  float         v8f;
typedef __attribute__((ext_vector_type(8)))  unsigned int  uint8v;

union FragCast  { uint8v u; bf16x16 b; };

static __device__ __forceinline__ unsigned short f32_to_bf16(float f) {
    union { float f; unsigned int u; } v; v.f = f;
    unsigned int u = v.u;
    unsigned int r = u + 0x7FFFu + ((u >> 16) & 1u);   // round-to-nearest-even
    if ((u & 0x7F800000u) == 0x7F800000u) r = u;       // keep Inf/NaN exponent
    return (unsigned short)(r >> 16);
}

// Generic shared pointer -> wave-relative LDS byte offset.
// ISA 10.2: LDS aperture maps addr[31:0] directly, so low 32 bits == offset.
static __device__ __forceinline__ unsigned lds_byte_off(const void* p) {
    return (unsigned)(unsigned long long)p;
}

// Async DMA-style copy: 16 bytes global -> LDS, tracked by ASYNCcnt.
static __device__ __forceinline__ void
async_copy_b128(unsigned lds_off, const void* gaddr) {
    asm volatile("global_load_async_to_lds_b128 %0, %1, off"
                 :: "v"(lds_off), "v"(gaddr) : "memory");
}
static __device__ __forceinline__ void wait_async0() {
    asm volatile("s_wait_asynccnt 0x0" ::: "memory");
}

// ---------------------------------------------------------------------------
// dst_bf16[i] = bf16( relu? max(src,0) : src )
// ---------------------------------------------------------------------------
__global__ void cvt_f32_bf16(const float* __restrict__ src,
                             unsigned short* __restrict__ dst, int n4, int relu) {
    int i = blockIdx.x * blockDim.x + threadIdx.x;
    if (i >= n4) return;
    float4 f = ((const float4*)src)[i];
    if (relu) {
        f.x = f.x > 0.f ? f.x : 0.f;  f.y = f.y > 0.f ? f.y : 0.f;
        f.z = f.z > 0.f ? f.z : 0.f;  f.w = f.w > 0.f ? f.w : 0.f;
    }
    ushort4 o;
    o.x = f32_to_bf16(f.x); o.y = f32_to_bf16(f.y);
    o.z = f32_to_bf16(f.z); o.w = f32_to_bf16(f.w);
    ((ushort4*)dst)[i] = o;
}

// ---------------------------------------------------------------------------
// One-time weight demotion + swizzle:  W[K][N] f32 -> Bsw[K/32][N][32] bf16.
// Panelized so a GEMM B tile (k-panel, 128 cols) is one contiguous 8KB blob
// that can be async-copied straight into LDS with zero repacking.
// grid = (N/64, K/32), block = 256.
// ---------------------------------------------------------------------------
__global__ void __launch_bounds__(256)
cvt_w_swizzle(const float* __restrict__ W, unsigned short* __restrict__ Bsw,
              int N, int K) {
    __shared__ unsigned short tile[64][32 + 8];   // 80B rows, 16B aligned
    const int tid = threadIdx.x;
    const int p   = blockIdx.y;          // k-panel index
    const int n0  = blockIdx.x * 64;

    // coalesced read: 8 consecutive n of one k row
    const int kk = tid >> 3;             // 0..31
    const int ng = (tid & 7) * 8;        // 0..56
    const float4* src = (const float4*)(W + (size_t)(p * 32 + kk) * N + n0 + ng);
    float4 x = src[0], y = src[1];
    float f[8] = {x.x, x.y, x.z, x.w, y.x, y.y, y.z, y.w};
    #pragma unroll
    for (int j = 0; j < 8; ++j)
        tile[ng + j][kk] = f32_to_bf16(f[j]);
    __syncthreads();

    // coalesced write: 8 contiguous bf16 of one n row
    const int n = tid >> 1;
    const int c = (tid & 1) * 16;
    *(uint4*)(Bsw + ((size_t)p * N + n0 + n) * 32 + c) = *(const uint4*)&tile[n][c];
}

// ---------------------------------------------------------------------------
// t[v,:] = (1 + eps[layer]) * h[v,:]
// ---------------------------------------------------------------------------
__global__ void scale_self(const float* __restrict__ h,
                           const float* __restrict__ eps, int layer,
                           float* __restrict__ t, int n) {
    int i = blockIdx.x * blockDim.x + threadIdx.x;
    float s = 1.0f + eps[layer];
    if (i < n) t[i] = s * h[i];
}

// ---------------------------------------------------------------------------
// t[dst,:] += h[src,:]  (segment-sum via f32 atomics; t is L2-resident, 20MB)
// ---------------------------------------------------------------------------
__global__ void edge_scatter(const float* __restrict__ h,
                             const int* __restrict__ edges,  // [2][N_EDGES]
                             float* __restrict__ t) {
    const int CH  = DHID / 4;
    int idx = blockIdx.x * blockDim.x + threadIdx.x;
    int e   = idx / CH;
    int d4  = idx - e * CH;
    if (e >= NN_EDGES) return;
    int src = edges[e];
    int dst = edges[NN_EDGES + e];
    const float4 v = ((const float4*)(h + (size_t)src * DHID))[d4];
    float* p = t + (size_t)dst * DHID + (size_t)d4 * 4;
    atomicAdd(p + 0, v.x);
    atomicAdd(p + 1, v.y);
    atomicAdd(p + 2, v.z);
    atomicAdd(p + 3, v.w);
}

// ---------------------------------------------------------------------------
// abf[i] = bf16( 0.5*(a[i]+b[i]) )  — fused layer-mean + demotion
// ---------------------------------------------------------------------------
__global__ void mean2_bf16(const float* __restrict__ a, const float* __restrict__ b,
                           unsigned short* __restrict__ dst, int n4) {
    int i = blockIdx.x * blockDim.x + threadIdx.x;
    if (i >= n4) return;
    float4 fa = ((const float4*)a)[i];
    float4 fb = ((const float4*)b)[i];
    ushort4 o;
    o.x = f32_to_bf16(0.5f * (fa.x + fb.x));
    o.y = f32_to_bf16(0.5f * (fa.y + fb.y));
    o.z = f32_to_bf16(0.5f * (fa.z + fb.z));
    o.w = f32_to_bf16(0.5f * (fa.w + fb.w));
    ((ushort4*)dst)[i] = o;
}

// ---------------------------------------------------------------------------
// C[M,N](f32) = f(Abf16[M,K] @ W + bias) via v_wmma_f32_16x16x32_bf16,
// where W is pre-swizzled Bsw[K/32][N][32] bf16.
//   resid != nullptr : C = relu(acc+bias) + relu(resid[row,col])   (N == K == D)
//   resid == nullptr : C = acc + bias
// Block: 256 threads = 8 waves; block tile 128x128; K-step 32; LDS ping-pong.
// BOTH tiles staged with async global->LDS copies (ASYNCcnt) — no VGPR
// round-trip, no repacking in the hot loop.
// Requires K % 32 == 0, N % 128 == 0; M ragged (A source row clamped — rows
// >= M feed only C rows >= M, which are never stored).
// ---------------------------------------------------------------------------
#define BM 128
#define BN 128
#define BK 32
#define KPAD 8   // shorts; LDS row stride 40 shorts = 80 B (16 B aligned)

__global__ void __launch_bounds__(256)
gemm_bf16_wmma(const unsigned short* __restrict__ A,
               const unsigned short* __restrict__ Bsw,
               const float* __restrict__ bias, const float* __restrict__ resid,
               float* __restrict__ C, int M, int N, int K) {
    __shared__ unsigned short lds_a[2][BM][BK + KPAD];   // [buf][m][k]
    __shared__ unsigned short lds_b[2][BN][BK + KPAD];   // [buf][n][k]

    const int tid  = threadIdx.x;
    const int lane = tid & 31;
    const int wave = tid >> 5;       // 0..7 : 16-row stripe of the block tile
    const int half = lane >> 4;
    const int l16  = lane & 15;

    const int m0 = blockIdx.y * BM;
    const int n0 = blockIdx.x * BN;

    v8f acc[8];
    #pragma unroll
    for (int j = 0; j < 8; ++j)
        acc[j] = (v8f){0.f,0.f,0.f,0.f,0.f,0.f,0.f,0.f};

    // staging map: thread owns 32B of one row in each tile
    const int t_row = tid >> 1;                     // 0..127
    const int t_col = (tid & 1) * 16;               // 0 or 16 (shorts)
    const int a_gr  = (m0 + t_row < M) ? (m0 + t_row) : (M - 1);   // clamp

    const unsigned short* a_src = A + (size_t)a_gr * K + t_col;           // +k0
    const unsigned short* b_src = Bsw + ((size_t)(n0 + t_row)) * 32 + t_col; // +panel*N*32

    const int nsteps = K / BK;

    // ---- stage one tile (both A and B) into buffer `buf` for k-step `s` ----
    auto stage = [&](int buf, int s) {
        const unsigned short* ap = a_src + s * BK;
        async_copy_b128(lds_byte_off(&lds_a[buf][t_row][t_col]),     ap);
        async_copy_b128(lds_byte_off(&lds_a[buf][t_row][t_col + 8]), ap + 8);
        const unsigned short* bp = b_src + (size_t)s * N * 32;
        async_copy_b128(lds_byte_off(&lds_b[buf][t_row][t_col]),     bp);
        async_copy_b128(lds_byte_off(&lds_b[buf][t_row][t_col + 8]), bp + 8);
    };

    // ---- prologue ----
    stage(0, 0);
    wait_async0();
    __syncthreads();

    for (int step = 0; step < nsteps; ++step) {
        const int cur = step & 1;
        const bool has_next = (step + 1) < nsteps;

        if (has_next) {
            stage(cur ^ 1, step + 1);              // overlaps with WMMAs below
            if (step + 2 < nsteps)                 // warm L2 for the tile after
                __builtin_prefetch(b_src + (size_t)(step + 2) * N * 32, 0, 3);
        }

        // ---- compute on buffer cur ----
        FragCast ac;
        const int am = wave * 16 + l16;
        #pragma unroll
        for (int p = 0; p < 8; ++p) {
            int kk = (p < 4) ? (half * 8 + 2 * p)
                             : (16 + half * 8 + 2 * (p - 4));
            ac.u[p] = *(const unsigned int*)&lds_a[cur][am][kk];
        }
        bf16x16 afrag = ac.b;

        #pragma unroll
        for (int j = 0; j < 8; ++j) {
            FragCast bc;
            const int bn = j * 16 + l16;
            #pragma unroll
            for (int p = 0; p < 8; ++p)
                bc.u[p] = *(const unsigned int*)&lds_b[cur][bn][(p < 4) ? (half * 8 + 2 * p) : (16 + half * 8 + 2 * (p - 4))];
            acc[j] = __builtin_amdgcn_wmma_f32_16x16x32_bf16(
                         false, afrag, false, bc.b,
                         (short)0, acc[j], false, false);
        }

        if (has_next) wait_async0();
        __syncthreads();
    }

    // ---- epilogue: C/D layout lane -> (M = 8*half + r, N = l16) ----
    #pragma unroll
    for (int j = 0; j < 8; ++j) {
        const int col  = n0 + j * 16 + l16;
        const float bc = bias ? bias[col] : 0.f;
        #pragma unroll
        for (int r = 0; r < 8; ++r) {
            int row = m0 + wave * 16 + half * 8 + r;
            if (row < M) {
                float v = acc[j][r] + bc;
                if (resid) {
                    float rr = resid[(size_t)row * N + col];
                    rr = rr > 0.f ? rr : 0.f;
                    v = (v > 0.f ? v : 0.f) + rr;
                }
                C[(size_t)row * N + col] = v;
            }
        }
    }
}

// ---------------------------------------------------------------------------
extern "C" void kernel_launch(void* const* d_in, const int* in_sizes, int n_in,
                              void* d_out, int out_size, void* d_ws, size_t ws_size,
                              hipStream_t stream) {
    const float* x     = (const float*)d_in[0];
    const int*   edges = (const int*)  d_in[1];
    const float* eps   = (const float*)d_in[2];
    const float* gin_w = (const float*)d_in[3];
    const float* gin_b = (const float*)d_in[4];
    const float* res_w = (const float*)d_in[5];
    const float* res_b = (const float*)d_in[6];
    const float* out_w = (const float*)d_in[7];
    const float* out_b = (const float*)d_in[8];
    float* logits = (float*)d_out;

    const size_t nd = (size_t)NN_NODES * DHID;   // 5.12M
    const size_t dd = (size_t)DHID * DHID;       // 262144
    const size_t dl = (size_t)DHID * NN_LABELS;  // 4.19M

    // f32 scratch
    float* t    = (float*)d_ws;       // agg buffer
    float* h1   = t    + nd;          // post-GIN-linear hidden
    float* hid0 = h1   + nd;          // layer-0 output
    float* hid1 = hid0 + nd;          // layer-1 output
    // bf16 scratch (weights stored pre-swizzled as [K/32][N][32] panels)
    unsigned short* abf    = (unsigned short*)(hid1 + nd);  // activations, nd
    unsigned short* wb_gin = abf    + nd;                   // 2*dd
    unsigned short* wb_res = wb_gin + 2 * dd;               // 2*dd
    unsigned short* wb_out = wb_res + 2 * dd;               // dl
    // footprint ~= 103 MB

    dim3 blk(256);
    const int ew_elems  = (int)nd;
    const int ew_blocks = (ew_elems + 255) / 256;
    const int sc_blocks = (NN_EDGES * (DHID / 4) + 255) / 256;
    const int nd4 = (int)(nd / 4);
    const int cnd = (nd4 + 255) / 256;

    dim3 g_wdd(DHID / 64,      DHID / 32);                 // weight swizzle D x D
    dim3 g_wout(NN_LABELS / 64, DHID / 32);                // weight swizzle D x L
    dim3 g_dd (DHID / BN,      (NN_NODES + BM - 1) / BM);  // 4 x 79
    dim3 g_out(NN_LABELS / BN, (NN_NODES + BM - 1) / BM);  // 64 x 79

    // ---- one-time weight demotion + panel swizzle ----
    cvt_w_swizzle<<<g_wdd,  blk, 0, stream>>>(gin_w,      wb_gin,      DHID, DHID);
    cvt_w_swizzle<<<g_wdd,  blk, 0, stream>>>(gin_w + dd, wb_gin + dd, DHID, DHID);
    cvt_w_swizzle<<<g_wdd,  blk, 0, stream>>>(res_w,      wb_res,      DHID, DHID);
    cvt_w_swizzle<<<g_wdd,  blk, 0, stream>>>(res_w + dd, wb_res + dd, DHID, DHID);
    cvt_w_swizzle<<<g_wout, blk, 0, stream>>>(out_w,      wb_out,      NN_LABELS, DHID);

    // ---------------- layer 0 ----------------
    scale_self  <<<ew_blocks, blk, 0, stream>>>(x, eps, 0, t, ew_elems);
    edge_scatter<<<sc_blocks, blk, 0, stream>>>(x, edges, t);
    cvt_f32_bf16<<<cnd, blk, 0, stream>>>(t, abf, nd4, 0);
    gemm_bf16_wmma<<<g_dd, blk, 0, stream>>>(abf, wb_gin, gin_b, nullptr, h1,
                                             NN_NODES, DHID, DHID);
    cvt_f32_bf16<<<cnd, blk, 0, stream>>>(h1, abf, nd4, 1);   // A := relu(h1)
    gemm_bf16_wmma<<<g_dd, blk, 0, stream>>>(abf, wb_res, res_b, h1, hid0,
                                             NN_NODES, DHID, DHID);
    // ---------------- layer 1 ----------------
    scale_self  <<<ew_blocks, blk, 0, stream>>>(hid0, eps, 1, t, ew_elems);
    edge_scatter<<<sc_blocks, blk, 0, stream>>>(hid0, edges, t);
    cvt_f32_bf16<<<cnd, blk, 0, stream>>>(t, abf, nd4, 0);
    gemm_bf16_wmma<<<g_dd, blk, 0, stream>>>(abf, wb_gin + dd, gin_b + DHID,
                                             nullptr, h1, NN_NODES, DHID, DHID);
    cvt_f32_bf16<<<cnd, blk, 0, stream>>>(h1, abf, nd4, 1);
    gemm_bf16_wmma<<<g_dd, blk, 0, stream>>>(abf, wb_res + dd, res_b + DHID,
                                             h1, hid1, NN_NODES, DHID, DHID);
    // ---------------- pool + projection ----------------
    mean2_bf16<<<cnd, blk, 0, stream>>>(hid0, hid1, abf, nd4);
    gemm_bf16_wmma<<<g_out, blk, 0, stream>>>(abf, wb_out, out_b, nullptr, logits,
                                              NN_NODES, NN_LABELS, DHID);
}
